// Segmentor_34265249088135
// MI455X (gfx1250) — matrix operand order repaired and checked by reading, not compiled, not run on hardware
//
#include <hip/hip_runtime.h>
#include <hip/hip_bf16.h>

// ---------------------------------------------------------------------------
// CDNA5 (gfx1250) implementation of the Segmentor reference:
//   backbone: 3x (conv3x3 + BN + ReLU) as WMMA implicit GEMM, f16 data / f32 acc
//   layer 2 fuses the 1x1 feature projection (64 -> 16 ch) via extra WMMAs
//   head: per-patch gather + pos-code + BN/ReLU + stride-2 transposed conv +
//         sigmoid, plus integer instance coords.
// Intermediates are f16 in d_ws.  Interior blocks of the f16-input layers
// stage their halo with GLOBAL_LOAD_ASYNC_TO_LDS_B128 (ASYNCcnt path).
// ---------------------------------------------------------------------------

typedef _Float16 h8   __attribute__((ext_vector_type(8)));
typedef _Float16 v16h __attribute__((ext_vector_type(16)));
typedef float    v8f  __attribute__((ext_vector_type(8)));

#define BN_EPS 1e-3f

static constexpr int Bb = 4, Hh = 512, Ww = 512;
static constexpr int TILE = 128;          // output pixels per block (1 row span)
static constexpr int PS = 48, CROP = 96;  // patch size / instance crop

// Raw LDS byte offset of a (generic) pointer into shared memory: addrspacecast
// to AS(3) then ptrtoint.  (A plain (size_t) cast would yield the flat-aperture
// address, which is NOT what VDST of an async-to-LDS instruction wants.)
__device__ __forceinline__ unsigned lds_offset(const void* p) {
    return (unsigned)(unsigned long long)
           (__attribute__((address_space(3))) const char*)p;
}

// ---------------------------------------------------------------------------
// conv3x3 + BN + ReLU (implicit GEMM, WMMA f16 -> f32), COUT = 64.
// Block: 256 threads = 8 wave32; each wave owns 16 pixels x 64 channels.
// ---------------------------------------------------------------------------
template<int CIN, bool IN_F32, bool FUSE>
__global__ void __launch_bounds__(256)
conv3x3_bn_relu(const void* __restrict__ src_, _Float16* __restrict__ dst,
                const float* __restrict__ Wt,
                const float* __restrict__ gamma, const float* __restrict__ beta,
                const float* __restrict__ mean,  const float* __restrict__ var,
                const float* __restrict__ fw,    _Float16* __restrict__ fmap)
{
    constexpr int KCH = CIN / 32;               // K chunks of 32
    constexpr int NW  = 9 * KCH * 4 * 512;      // weight-fragment halves
    constexpr int NI  = 3 * (TILE + 2) * CIN;   // input-halo halves

    extern __shared__ __align__(32) char smem[];
    _Float16* wlds  = (_Float16*)smem;          // [t][cc][nt][lane][16]
    _Float16* inT   = wlds + NW;                // [3][130][CIN]
    _Float16* ffrag = inT + NI;                 // FUSE: [2][lane][16]
    _Float16* xlds  = ffrag + (FUSE ? 1024 : 0);// FUSE: [wave][16][64]

    const int tid = threadIdx.x;
    const int xb  = blockIdx.x & (Ww / TILE - 1);
    const int y   = (blockIdx.x >> 2) & (Hh - 1);
    const int b   = blockIdx.x >> 11;
    const int x0  = xb * TILE;

    // -- speculative L2 prefetch of the neighboring tile (global_prefetch_b8)
    if (tid == 0 && x0 + TILE < Ww) {
        size_t off = ((size_t)(b * Hh + y) * Ww + x0 + TILE) * CIN;
        __builtin_prefetch((const char*)src_ + off * (IN_F32 ? 4 : 2), 0, 1);
    }

    // -- stage weights into WMMA B-fragment layout:
    //    frag[lane][j] = W[k = cc*32 + (j>>3)*16 + (lane>>4)*8 + (j&7)][n = nt*16 + (lane&15)]
    for (int i = tid; i < NW; i += 256) {
        int j  = i & 15;
        int ln = (i >> 4) & 31;
        int nt = (i >> 9) & 3;
        int r  = i >> 11;
        int cc = r % KCH;
        int t  = r / KCH;
        int k  = cc * 32 + (j >> 3) * 16 + (ln >> 4) * 8 + (j & 7);
        wlds[i] = (_Float16)Wt[(size_t)(t * CIN + k) * 64 + nt * 16 + (ln & 15)];
    }
    if constexpr (FUSE) {
        for (int i = tid; i < 1024; i += 256) {
            int j  = i & 15;
            int ln = (i >> 4) & 31;
            int cc = i >> 9;
            int k  = cc * 32 + (j >> 3) * 16 + (ln >> 4) * 8 + (j & 7);
            ffrag[i] = (_Float16)fw[k * 16 + (ln & 15)];
        }
    }

    // -- stage input halo ([3][130][CIN] f16, 'SAME' zero padding).
    bool async_ok = false;
    if constexpr (!IN_F32) {
        // Interior block: halo rows are 3 fully in-range contiguous byte runs
        // -> DMA them straight into LDS (no VGPR bounce), tracked by ASYNCcnt.
        if (y > 0 && y < Hh - 1 && x0 > 0 && x0 + TILE < Ww) {
            async_ok = true;
            constexpr int rowBytes = (TILE + 2) * CIN * 2;   // 16640 (CIN=64)
            constexpr int nChunks  = 3 * rowBytes / 16;
            const char* src = (const char*)src_;
            const unsigned ldsBase = lds_offset(inT);
            for (int c = tid; c < nChunks; c += 256) {
                int o  = c * 16;
                int ry = o / rowBytes;
                int ro = o - ry * rowBytes;
                const char* gp = src +
                    (((size_t)(b * Hh + (y - 1 + ry)) * Ww + (x0 - 1)) * CIN) * 2 + ro;
                unsigned ldst = ldsBase + (unsigned)o;
                asm volatile("global_load_async_to_lds_b128 %0, %1, off"
                             :: "v"(ldst), "v"((unsigned long long)gp)
                             : "memory");
            }
            asm volatile("s_wait_asynccnt 0" ::: "memory");
        }
    }
    if (!async_ok) {
        for (int i = tid; i < NI; i += 256) {
            int c  = i % CIN;
            int x  = (i / CIN) % (TILE + 2);
            int ry = i / (CIN * (TILE + 2));
            int gy = y + ry - 1;
            int gx = x0 + x - 1;
            float v = 0.f;
            if (gy >= 0 && gy < Hh && gx >= 0 && gx < Ww) {
                size_t off = ((size_t)(b * Hh + gy) * Ww + gx) * CIN + c;
                v = IN_F32 ? ((const float*)src_)[off]
                           : (float)((const _Float16*)src_)[off];
            }
            inT[i] = (_Float16)v;
        }
    }
    __syncthreads();

    const int wave = tid >> 5;        // wave32
    const int lane = tid & 31;
    const int m    = lane & 15;       // row within wave tile / column N
    const int hsel = lane >> 4;       // half-select per WMMA VGPR layouts
    const int poff = wave * 16 + m;   // pixel offset within the 128-px block

    v8f acc[4] = {};
    #pragma unroll
    for (int t = 0; t < 9; ++t) {
        const int ky = t / 3, kx = t % 3;
        const _Float16* arow = inT + (ky * (TILE + 2) + poff + kx) * CIN;
        #pragma unroll
        for (int cc = 0; cc < KCH; ++cc) {
            // A fragment (16x32 f16): lane holds K in [h*8,h*8+8) and [16+h*8,..)
            h8 lo = *(const h8*)(arow + cc * 32 + hsel * 8);
            h8 hi = *(const h8*)(arow + cc * 32 + 16 + hsel * 8);
            v16h a = __builtin_shufflevector(lo, hi,
                     0,1,2,3,4,5,6,7,8,9,10,11,12,13,14,15);
            const _Float16* bp = wlds + (size_t)((t * KCH + cc) * 4) * 512 + lane * 16;
            // hoist all 4 N-tile B fragments so ds latency overlaps WMMA issue
            v16h bf0 = *(const v16h*)(bp);
            v16h bf1 = *(const v16h*)(bp + 512);
            v16h bf2 = *(const v16h*)(bp + 1024);
            v16h bf3 = *(const v16h*)(bp + 1536);
            acc[0] = __builtin_amdgcn_wmma_f32_16x16x32_f16(
                         false, a, false, bf0, (short)0, acc[0], false, false);
            acc[1] = __builtin_amdgcn_wmma_f32_16x16x32_f16(
                         false, a, false, bf1, (short)0, acc[1], false, false);
            acc[2] = __builtin_amdgcn_wmma_f32_16x16x32_f16(
                         false, a, false, bf2, (short)0, acc[2], false, false);
            acc[3] = __builtin_amdgcn_wmma_f32_16x16x32_f16(
                         false, a, false, bf3, (short)0, acc[3], false, false);
        }
    }

    // -- BN (folded) + ReLU, per C/D layout: ch = nt*16 + m, row = r + 8*hsel
    const size_t pixBase = ((size_t)(b * Hh + y)) * Ww + x0 + wave * 16;
    #pragma unroll
    for (int nt = 0; nt < 4; ++nt) {
        int ch   = nt * 16 + m;
        float s  = gamma[ch] * rsqrtf(var[ch] + BN_EPS);
        float bs = beta[ch] - mean[ch] * s;
        #pragma unroll
        for (int r = 0; r < 8; ++r) {
            int row = r + hsel * 8;
            float v = fmaxf(acc[nt][r] * s + bs, 0.f);
            if constexpr (FUSE)
                xlds[wave * 1024 + row * 64 + ch] = (_Float16)v;   // same-wave LDS
            else
                dst[(pixBase + row) * 64 + ch] = (_Float16)v;
        }
    }

    if constexpr (FUSE) {
        // 1x1 projection 64 -> 16 through per-wave LDS (no barrier: same wave)
        v8f fa = {};
        const _Float16* xr = xlds + wave * 1024 + m * 64;
        #pragma unroll
        for (int cc = 0; cc < 2; ++cc) {
            h8 lo = *(const h8*)(xr + cc * 32 + hsel * 8);
            h8 hi = *(const h8*)(xr + cc * 32 + 16 + hsel * 8);
            v16h a = __builtin_shufflevector(lo, hi,
                     0,1,2,3,4,5,6,7,8,9,10,11,12,13,14,15);
            v16h bf = *(const v16h*)(ffrag + cc * 512 + lane * 16);
            fa = __builtin_amdgcn_wmma_f32_16x16x32_f16(
                     false, a, false, bf, (short)0, fa, false, false);
        }
        #pragma unroll
        for (int r = 0; r < 8; ++r) {
            int row = r + hsel * 8;
            fmap[(pixBase + row) * 16 + m] = (_Float16)fa[r];
        }
    }
}

// ---------------------------------------------------------------------------
// Per-patch head: gather 48x48x16 (clamped), add position code, BN+ReLU,
// stride-2 3x3 transposed conv to 96x96, sigmoid; emit instance coords.
// ---------------------------------------------------------------------------
__global__ void __launch_bounds__(256)
head_kernel(const _Float16* __restrict__ fmap, const float* __restrict__ loc,
            const float* __restrict__ cw,
            const float* __restrict__ mg, const float* __restrict__ mb,
            const float* __restrict__ mm, const float* __restrict__ mv,
            const float* __restrict__ ow, const float* __restrict__ ob,
            float* __restrict__ out, int* __restrict__ coords)
{
    extern __shared__ __align__(32) char smem[];
    _Float16* p  = (_Float16*)smem;            // [48][48][16] post-BN/ReLU
    float*    wl = (float*)(p + PS * PS * 16); // [3][3][16] out_w

    const int bn = blockIdx.x;
    const float ly = loc[bn * 2 + 0] * (float)Hh;
    const float lx = loc[bn * 2 + 1] * (float)Ww;
    const int i0y = (int)ly - PS / 2;          // unclipped top-left (for coords)
    const int i0x = (int)lx - PS / 2;
    const int b   = bn >> 8;

    for (int i = threadIdx.x; i < 144; i += 256) wl[i] = ow[i];

    for (int t = threadIdx.x; t < PS * PS; t += 256) {
        int py = t / PS, px = t % PS;
        int iy = min(max(i0y + py, 0), Hh - 1);
        int ix = min(max(i0x + px, 0), Ww - 1);
        const _Float16* f = fmap + ((size_t)(b * Hh + iy) * Ww + ix) * 16;
        float ry = (py - PS * 0.5f) / (float)PS;   // pos code, recomputed
        float rx = (px - PS * 0.5f) / (float)PS;
        #pragma unroll
        for (int c = 0; c < 16; ++c) {
            float v = (float)f[c] + ry * cw[c] + rx * cw[16 + c];
            float s = mg[c] * rsqrtf(mv[c] + BN_EPS);
            v = v * s + (mb[c] - mm[c] * s);
            p[t * 16 + c] = (_Float16)fmaxf(v, 0.f);
        }
    }
    __syncthreads();

    const float bias = ob[0];
    for (int o = threadIdx.x; o < CROP * CROP; o += 256) {
        int oy = o / CROP, ox = o % CROP;
        float acc = bias;
        // transposed conv (stride 2, k 3, SAME): o = 2*i + k - 1
        #pragma unroll
        for (int ky = 0; ky < 3; ++ky) {
            int ty = oy + 1 - ky;
            if (ty < 0 || (ty & 1)) continue;
            int iy = ty >> 1;
            if (iy >= PS) continue;
            #pragma unroll
            for (int kx = 0; kx < 3; ++kx) {
                int tx = ox + 1 - kx;
                if (tx < 0 || (tx & 1)) continue;
                int ix = tx >> 1;
                if (ix >= PS) continue;
                const _Float16* pp = p + (iy * PS + ix) * 16;
                const float*    wp = wl + (ky * 3 + kx) * 16;
                #pragma unroll
                for (int c = 0; c < 16; ++c) acc += (float)pp[c] * wp[c];
            }
        }
        size_t oo = (size_t)bn * (CROP * CROP) + o;
        out[oo] = 1.f / (1.f + expf(-acc));
        coords[oo * 2 + 0] = i0y * 2 + oy;     // unclipped idx * SCALE + mesh
        coords[oo * 2 + 1] = i0x * 2 + ox;
    }
}

// ---------------------------------------------------------------------------
extern "C" void kernel_launch(void* const* d_in, const int* in_sizes, int n_in,
                              void* d_out, int out_size, void* d_ws, size_t ws_size,
                              hipStream_t stream)
{
    (void)in_sizes; (void)n_in; (void)out_size; (void)ws_size;
    const float* hr  = (const float*)d_in[0];   // [4,512,512,32]
    const float* loc = (const float*)d_in[1];   // [4,256,2]
    const float* w0  = (const float*)d_in[2];
    const float* w1  = (const float*)d_in[3];
    const float* w2  = (const float*)d_in[4];
    const float* g   = (const float*)d_in[5];   // [3,64]
    const float* bt  = (const float*)d_in[6];
    const float* mn  = (const float*)d_in[7];
    const float* vr  = (const float*)d_in[8];
    const float* fw  = (const float*)d_in[9];   // [64,16]
    const float* cw  = (const float*)d_in[10];  // [2,16]
    // d_in[11] = pos (recomputed analytically in head_kernel)
    const float* mg  = (const float*)d_in[12];
    const float* mb  = (const float*)d_in[13];
    const float* mm  = (const float*)d_in[14];
    const float* mv  = (const float*)d_in[15];
    const float* ow  = (const float*)d_in[16];  // [3,3,16,1]
    const float* ob  = (const float*)d_in[17];

    // f16 ping-pong in workspace; fmap reuses ws0 (conv0 output dead by then)
    _Float16* ws0 = (_Float16*)d_ws;
    _Float16* ws1 = ws0 + (size_t)Bb * Hh * Ww * 64;   // +128 MB

    const int convBlocks = Bb * Hh * (Ww / TILE);      // 8192
    size_t smem32  = (size_t)(9 * 1 * 4 * 512 + 3 * (TILE + 2) * 32) * 2;
    size_t smem64  = (size_t)(9 * 2 * 4 * 512 + 3 * (TILE + 2) * 64) * 2;
    size_t smem64f = smem64 + (size_t)(1024 + 8 * 1024) * 2;

    conv3x3_bn_relu<32, true, false><<<convBlocks, 256, smem32, stream>>>(
        hr, ws0, w0, g + 0, bt + 0, mn + 0, vr + 0, nullptr, nullptr);
    conv3x3_bn_relu<64, false, false><<<convBlocks, 256, smem64, stream>>>(
        ws0, ws1, w1, g + 64, bt + 64, mn + 64, vr + 64, nullptr, nullptr);
    conv3x3_bn_relu<64, false, true><<<convBlocks, 256, smem64f, stream>>>(
        ws1, nullptr, w2, g + 128, bt + 128, mn + 128, vr + 128, fw, ws0);

    float* out   = (float*)d_out;                              // [4,256,96,96,1]
    int* coords  = (int*)(out + (size_t)Bb * 256 * CROP * CROP); // int32 coords
    size_t smemH = (size_t)PS * PS * 16 * 2 + 144 * 4;
    head_kernel<<<Bb * 256, 256, smemH, stream>>>(
        ws0, loc, cw, mg, mb, mm, mv, ow, ob, out, coords);
}
